// MultiHeadAttentionUnit_46394236731811
// MI455X (gfx1250) — matrix operand order repaired and checked by reading, not compiled, not run on hardware
//
#include <hip/hip_runtime.h>

// MI455X (gfx1250) wave32 WMMA implementation of the attention unit.
// bf16 matrix ops with f32 accumulation; streaming recompute of scores
// (S x S x H scores never materialized: 536MB > L2; Q/K/V bf16 ~12.6MB fits L2).
// All per-lane operand gathers are contiguous 16B chunks -> b128 loads.

#define B_    2
#define S_    4096
#define DIM_  768
#define G_    256
#define H_    4
#define DK_   64
#define SCALE_ 0.125f   // 1/sqrt(64)

typedef __attribute__((ext_vector_type(16))) __bf16 v16bf;
typedef __attribute__((ext_vector_type(8)))  __bf16 v8bf;
typedef __attribute__((ext_vector_type(8)))  float  v8f;
typedef __attribute__((ext_vector_type(4)))  float  v4f;

// A-matrix (16x32 bf16): lane holds row M=lane%16; K chunks [hi*8..+7] and [16+hi*8..+7]
// B-matrix (32x16 bf16): lane holds col N=lane%16; K chunks [hi*8..+7] and [16+hi*8..+7]
// C/D (16x16 f32):       lane holds col N=lane%16; VGPR r -> row M = r + 8*hi

__device__ __forceinline__ v16bf cat16(v8bf a, v8bf b) {
  return __builtin_shufflevector(a, b, 0,1,2,3,4,5,6,7,8,9,10,11,12,13,14,15);
}
__device__ __forceinline__ v8bf ld8(const __bf16* p) { return *(const v8bf*)p; }
__device__ __forceinline__ v8bf cvt8(const float* p) {   // 2x float4 -> 8x bf16
  v4f c0 = *(const v4f*)p;
  v4f c1 = *(const v4f*)(p + 4);
  v8bf r;
  r[0]=(__bf16)c0[0]; r[1]=(__bf16)c0[1]; r[2]=(__bf16)c0[2]; r[3]=(__bf16)c0[3];
  r[4]=(__bf16)c1[0]; r[5]=(__bf16)c1[1]; r[6]=(__bf16)c1[2]; r[7]=(__bf16)c1[3];
  return r;
}
// bf16 A-tile / B-tile loaders (row = per-lane row pointer, kc = 32-wide K chunk)
__device__ __forceinline__ v16bf ldA(const __bf16* row, int kc, int hi) {
  return cat16(ld8(row + kc + hi * 8), ld8(row + kc + 16 + hi * 8));
}
__device__ __forceinline__ v16bf ldB(const __bf16* row, int kc, int hi) {
  return cat16(ld8(row + kc + hi * 8), ld8(row + kc + 16 + hi * 8));
}
__device__ __forceinline__ v16bf ldAf(const float* row, int kc, int hi) {
  return cat16(cvt8(row + kc + hi * 8), cvt8(row + kc + 16 + hi * 8));
}
__device__ __forceinline__ v16bf ldBf(const float* row, int kc, int hi) {
  return cat16(cvt8(row + kc + hi * 8), cvt8(row + kc + 16 + hi * 8));
}

// order-preserving float <-> uint key (atomicMax over floats incl. negatives)
__device__ __forceinline__ unsigned fkey(float f) {
  unsigned b = __float_as_uint(f);
  return (b & 0x80000000u) ? ~b : (b | 0x80000000u);
}
__device__ __forceinline__ float funkey(unsigned k) {
  unsigned b = (k & 0x80000000u) ? (k ^ 0x80000000u) : ~k;
  return __uint_as_float(b);
}

__device__ __forceinline__ v8f wmma_bf16(v16bf a, v16bf b, v8f c) {
  return __builtin_amdgcn_wmma_f32_16x16x32_bf16(
      /*neg_a=*/false, a, /*neg_b=*/false, b,
      /*c_mod=*/(short)0, c, /*reuse_a=*/false, /*reuse_b=*/false);
}

// ---------------------------------------------------------------- init stats
__global__ void k_init(unsigned* __restrict__ Mkey, float* __restrict__ Lsum, int n) {
  int i = blockIdx.x * blockDim.x + threadIdx.x;
  if (i < n) { Mkey[i] = 0u; Lsum[i] = 0.f; }
}

// ------------------------------------------------------------- QKV projection
// q[m,n] = sum_k x[m, mat*G+k] * W[n,k] + bias[n]
// Q,K stored bf16 [B][H][S][DK]; V stored transposed bf16 [B][H][DK][S]
__global__ void __launch_bounds__(256) k_qkv(
    const float* __restrict__ x,
    const float* __restrict__ wq, const float* __restrict__ bq,
    const float* __restrict__ wk, const float* __restrict__ bk,
    const float* __restrict__ wv, const float* __restrict__ bv,
    __bf16* __restrict__ Qb, __bf16* __restrict__ Kb, __bf16* __restrict__ Vt) {
  const int ROWT = (B_ * S_) / 16, COLT = G_ / 16;
  int wid  = blockIdx.x * (blockDim.x >> 5) + (threadIdx.x >> 5);
  int mat  = wid / (ROWT * COLT);
  int rem  = wid % (ROWT * COLT);
  int rowT = rem / COLT, colT = rem % COLT;
  int lane = threadIdx.x & 31, lo = lane & 15, hi = lane >> 4;

  const float* W    = (mat == 0) ? wq : (mat == 1) ? wk : wv;
  const float* bias = (mat == 0) ? bq : (mat == 1) ? bk : bv;

  int row0 = rowT * 16, n0 = colT * 16;
  const float* arow = x + (size_t)(row0 + lo) * DIM_ + mat * G_;
  const float* brow = W + (size_t)(n0 + lo) * G_;

  v8f acc = {};
  for (int kc = 0; kc < G_; kc += 32)
    acc = wmma_bf16(ldAf(arow, kc, hi), ldBf(brow, kc, hi), acc);

  float bia = bias[n0 + lo];
  int g = n0 + lo, h = g >> 6, dk = g & 63;
  int rowb = row0 + 8 * hi;                 // 8 consecutive output rows per lane
  int bb = rowb / S_, si0 = rowb % S_;
  if (mat == 2) {                           // V transposed: contiguous along S
    v8bf pv;
#pragma unroll
    for (int r = 0; r < 8; ++r) pv[r] = (__bf16)(acc[r] + bia);
    *(v8bf*)&Vt[((size_t)(bb * H_ + h) * DK_ + dk) * S_ + si0] = pv;
  } else {
    __bf16* dst = (mat == 0) ? Qb : Kb;
#pragma unroll
    for (int r = 0; r < 8; ++r)
      dst[((size_t)(bb * H_ + h) * S_ + si0 + r) * DK_ + dk] = (__bf16)(acc[r] + bia);
  }
}

// ------------------------------------------------- pass 1: per-column max_i s
__global__ void __launch_bounds__(256) k_colmax(
    const __bf16* __restrict__ Qb, const __bf16* __restrict__ Kb,
    unsigned* __restrict__ Mkey) {
  const int JT = S_ / 16;
  int wid = blockIdx.x * (blockDim.x >> 5) + (threadIdx.x >> 5);
  int bh = wid / JT, j0 = (wid % JT) * 16;
  int lane = threadIdx.x & 31, lo = lane & 15, hi = lane >> 4;
  const __bf16* Qp = Qb + (size_t)bh * S_ * DK_;
  const __bf16* krow = Kb + (size_t)bh * S_ * DK_ + (size_t)(j0 + lo) * DK_;

  v16bf bt0 = ldB(krow, 0, hi), bt1 = ldB(krow, 32, hi);  // B = K^T, fixed per wave

  float mmax = -3.0e38f;
  for (int i0 = 0; i0 < S_; i0 += 16) {
    const __bf16* qrow = Qp + (size_t)(i0 + lo) * DK_;
    v8f acc = {};
    acc = wmma_bf16(ldA(qrow, 0, hi),  bt0, acc);
    acc = wmma_bf16(ldA(qrow, 32, hi), bt1, acc);
#pragma unroll
    for (int r = 0; r < 8; ++r) mmax = fmaxf(mmax, acc[r] * SCALE_);
  }
  // lanes L and L+16 hold the same column j = L%16 (rows 0-7 vs 8-15)
  mmax = fmaxf(mmax, __shfl_xor(mmax, 16, 32));
  if (lane < 16) atomicMax(&Mkey[bh * S_ + j0 + lo], fkey(mmax));
}

// -------------------------------------------- pass 2: per-column sum_i exp(.)
__global__ void __launch_bounds__(256) k_colsum(
    const __bf16* __restrict__ Qb, const __bf16* __restrict__ Kb,
    const unsigned* __restrict__ Mkey, float* __restrict__ Lsum) {
  const int JT = S_ / 16;
  int wid = blockIdx.x * (blockDim.x >> 5) + (threadIdx.x >> 5);
  int bh = wid / JT, j0 = (wid % JT) * 16;
  int lane = threadIdx.x & 31, lo = lane & 15, hi = lane >> 4;
  const __bf16* Qp = Qb + (size_t)bh * S_ * DK_;
  const __bf16* krow = Kb + (size_t)bh * S_ * DK_ + (size_t)(j0 + lo) * DK_;

  v16bf bt0 = ldB(krow, 0, hi), bt1 = ldB(krow, 32, hi);

  float m = funkey(Mkey[bh * S_ + j0 + lo]);
  float ssum = 0.f;
  for (int i0 = 0; i0 < S_; i0 += 16) {
    const __bf16* qrow = Qp + (size_t)(i0 + lo) * DK_;
    v8f acc = {};
    acc = wmma_bf16(ldA(qrow, 0, hi),  bt0, acc);
    acc = wmma_bf16(ldA(qrow, 32, hi), bt1, acc);
#pragma unroll
    for (int r = 0; r < 8; ++r) ssum += __expf(acc[r] * SCALE_ - m);
  }
  ssum += __shfl_xor(ssum, 16, 32);
  if (lane < 16) atomicAdd(&Lsum[bh * S_ + j0 + lo], ssum);
}

// --------------------- pass 3: out[i] = sum_j P[i,j] * v[j]   (no LDS needed)
// Compute S^T tiles (K as A-matrix, Q as B-matrix): D lane = col i, rows = j.
// That D layout IS the A-matrix layout of P for the P*V WMMA:
//   pa[e]   (e<8)  = P[i=lane%16][j = jblk +      hi*8 + e] = D(jtile0)[r=e]
//   pa[8+e]        = P[i=lane%16][j = jblk + 16 + hi*8 + e] = D(jtile1)[r=e]
__global__ void __launch_bounds__(256) k_attn(
    const __bf16* __restrict__ Qb, const __bf16* __restrict__ Kb,
    const __bf16* __restrict__ Vt, const unsigned* __restrict__ Mkey,
    const float* __restrict__ Lsum, __bf16* __restrict__ Ob) {
  const int IT = S_ / 16;
  int wid = blockIdx.x * (blockDim.x >> 5) + (threadIdx.x >> 5);
  int bh = wid / IT, i0 = (wid % IT) * 16;
  int lane = threadIdx.x & 31, lo = lane & 15, hi = lane >> 4;
  const __bf16* Kp = Kb + (size_t)bh * S_ * DK_;
  const __bf16* Vp = Vt + (size_t)bh * DK_ * S_;
  const unsigned* Mp = Mkey + bh * S_;
  const float*    Lp = Lsum + bh * S_;

  // Q as B-matrix (dk x i), fixed per wave
  const __bf16* qrow = Qb + (size_t)bh * S_ * DK_ + (size_t)(i0 + lo) * DK_;
  v16bf qb0 = ldB(qrow, 0, hi), qb1 = ldB(qrow, 32, hi);

  v8f o[4] = {};    // 16 i-rows x 64 dk
  for (int j0 = 0; j0 < S_; j0 += 32) {
    v16bf pa;
#pragma unroll
    for (int half = 0; half < 2; ++half) {
      int jb = j0 + half * 16;
      const __bf16* krow = Kp + (size_t)(jb + lo) * DK_;
      v8f st = {};                               // st[r] = s[i=lo, j=jb+r+8*hi]
      st = wmma_bf16(ldA(krow, 0, hi),  qb0, st);
      st = wmma_bf16(ldA(krow, 32, hi), qb1, st);
#pragma unroll
      for (int r = 0; r < 8; ++r) {              // contiguous stat loads -> b128
        int j = jb + 8 * hi + r;
        float m  = funkey(Mp[j]);
        float rl = 1.0f / Lp[j];
        pa[half * 8 + r] = (__bf16)(__expf(st[r] * SCALE_ - m) * rl);
      }
    }
#pragma unroll
    for (int d = 0; d < 4; ++d) {                // B = V^T tile, contiguous j
      const __bf16* vrow = Vp + (size_t)(d * 16 + lo) * S_ + j0;
      o[d] = wmma_bf16(pa, ldB(vrow, 0, hi), o[d]);
    }
  }
  int bb = bh >> 2, h = bh & 3;
#pragma unroll
  for (int d = 0; d < 4; ++d)
#pragma unroll
    for (int r = 0; r < 8; ++r)
      Ob[((size_t)bb * S_ + i0 + r + 8 * hi) * G_ + h * 64 + d * 16 + lo] =
          (__bf16)o[d][r];
}

// --------------------------------------------------------- output projection
__global__ void __launch_bounds__(256) k_oproj(
    const __bf16* __restrict__ Ob, const float* __restrict__ wo,
    const float* __restrict__ bo, float* __restrict__ out) {
  const int COLT = G_ / 16;
  int wid  = blockIdx.x * (blockDim.x >> 5) + (threadIdx.x >> 5);
  int rowT = wid / COLT, colT = wid % COLT;
  int lane = threadIdx.x & 31, lo = lane & 15, hi = lane >> 4;
  int row0 = rowT * 16, n0 = colT * 16;
  const __bf16* arow = Ob + (size_t)(row0 + lo) * G_;
  const float*  brow = wo + (size_t)(n0 + lo) * G_;

  v8f acc = {};
  for (int kc = 0; kc < G_; kc += 32)
    acc = wmma_bf16(ldA(arow, kc, hi), ldBf(brow, kc, hi), acc);

  float bia = bo[n0 + lo];
#pragma unroll
  for (int r = 0; r < 8; ++r)
    out[(size_t)(row0 + r + 8 * hi) * G_ + n0 + lo] = acc[r] + bia;
}

// ----------------------------------------------------------------- dispatch
extern "C" void kernel_launch(void* const* d_in, const int* in_sizes, int n_in,
                              void* d_out, int out_size, void* d_ws, size_t ws_size,
                              hipStream_t stream) {
  const float* x  = (const float*)d_in[0];
  const float* wq = (const float*)d_in[1];
  const float* bq = (const float*)d_in[2];
  const float* wk = (const float*)d_in[3];
  const float* bk = (const float*)d_in[4];
  const float* wv = (const float*)d_in[5];
  const float* bv = (const float*)d_in[6];
  const float* wo = (const float*)d_in[7];
  const float* bo = (const float*)d_in[8];

  char* ws = (char*)d_ws;
  const size_t nQ = (size_t)B_ * H_ * S_ * DK_;      // 2,097,152 elems
  __bf16*   Qb   = (__bf16*)(ws);                    // 4 MB
  __bf16*   Kb   = (__bf16*)(ws + nQ * 2);           // 4 MB
  __bf16*   Vt   = (__bf16*)(ws + nQ * 4);           // 4 MB (transposed [dk][S])
  __bf16*   Ob   = (__bf16*)(ws + nQ * 6);           // 4 MB
  unsigned* Mkey = (unsigned*)(ws + nQ * 8);         // 128 KB
  float*    Lsum = (float*)(ws + nQ * 8 + (size_t)B_ * H_ * S_ * 4); // 128 KB
  const int nStat = B_ * H_ * S_;                    // 32768

  k_init  <<<(nStat + 255) / 256, 256, 0, stream>>>(Mkey, Lsum, nStat);
  k_qkv   <<<3 * ((B_ * S_) / 16) * (G_ / 16) / 8, 256, 0, stream>>>(
              x, wq, bq, wk, bk, wv, bv, Qb, Kb, Vt);
  k_colmax<<<(B_ * H_ * (S_ / 16)) / 8, 256, 0, stream>>>(Qb, Kb, Mkey);
  k_colsum<<<(B_ * H_ * (S_ / 16)) / 8, 256, 0, stream>>>(Qb, Kb, Mkey, Lsum);
  k_attn  <<<(B_ * H_ * (S_ / 16)) / 8, 256, 0, stream>>>(Qb, Kb, Vt, Mkey, Lsum, Ob);
  k_oproj <<<((B_ * S_) / 16) * (G_ / 16) / 8, 256, 0, stream>>>(Ob, wo, bo, (float*)d_out);
}
